// AdaptiveGraphAttention_6176162971961
// MI455X (gfx1250) — matrix operand order repaired and checked by reading, not compiled, not run on hardware
//
#include <hip/hip_runtime.h>
#include <hip/hip_bf16.h>
#include <math.h>

// Problem constants (from reference)
#define Bz 16
#define Nn 64
#define Ss 65
#define Dd 512
#define Hh 8
#define HDd 64

typedef __attribute__((ext_vector_type(16))) __bf16 v16bf;
typedef __attribute__((ext_vector_type(8)))  __bf16 v8bf;
typedef __attribute__((ext_vector_type(4)))  __bf16 v4bf;
typedef __attribute__((ext_vector_type(8)))  float  v8f;

// ---------------------------------------------------------------------------
// Elementwise f32 -> bf16 (4 elements/thread; n multiple of 4).
// ---------------------------------------------------------------------------
__global__ __launch_bounds__(256) void to_bf16(
    const float* __restrict__ X, __bf16* __restrict__ Y, int n4)
{
  int i = blockIdx.x * 256 + threadIdx.x;
  if (i < n4) {
    float4 v = *(const float4*)(X + (size_t)i * 4);
    v4bf o;
    o[0] = (__bf16)v.x; o[1] = (__bf16)v.y; o[2] = (__bf16)v.z; o[3] = (__bf16)v.w;
    *(v4bf*)(Y + (size_t)i * 4) = o;
  }
}

// ---------------------------------------------------------------------------
// 512x512 f32 row-major W[k][n]  ->  bf16 Wt[n][k]  (LDS-tiled transpose).
// ---------------------------------------------------------------------------
__global__ __launch_bounds__(256) void transpose_to_bf16(
    const float* __restrict__ W, __bf16* __restrict__ Wt)
{
  __shared__ float tile[32][33];
  int bx = blockIdx.x * 32;   // n block
  int by = blockIdx.y * 32;   // k block
  int tx = threadIdx.x, ty = threadIdx.y;          // 32 x 8
#pragma unroll
  for (int r = 0; r < 32; r += 8)
    tile[ty + r][tx] = W[(size_t)(by + ty + r) * Dd + bx + tx];
  __syncthreads();
#pragma unroll
  for (int r = 0; r < 32; r += 8)
    Wt[(size_t)(bx + ty + r) * Dd + by + tx] = (__bf16)tile[tx][ty + r];
}

// ---------------------------------------------------------------------------
// WMMA GEMM: Y[M,N] = Xb[M,K](bf16) @ Wt[N,K](bf16, transposed) (+ bias f32).
// One wave computes a 32x32 tile = 2x2 WMMAs; fragments are loaded as
// contiguous b128s in contiguous k order (k = 16*half + s), same for A and B:
//   A lane: two 8-elem runs  k = kb + 16*half (+0..7) and +8
//   B lane: one 16-elem run  k = kb + 16*half (+0..15)   (col = lane&15)
// M arbitrary (clamped loads, guarded stores; EXEC all-ones through WMMA).
// ---------------------------------------------------------------------------
__global__ __launch_bounds__(32) void wmma_gemm_bias(
    const __bf16* __restrict__ Xb, const __bf16* __restrict__ Wt,
    const float* __restrict__ bias, float* __restrict__ Y,
    int M, int N, int K)
{
  const int lane = threadIdx.x & 31;
  const int mlo  = lane & 15;
  const int half = lane >> 4;
  const int nt0  = blockIdx.x * 32;
  const int mt0  = blockIdx.y * 32;

  int arow0 = mt0 + mlo;        if (arow0 >= M) arow0 = M - 1;
  int arow1 = mt0 + 16 + mlo;   if (arow1 >= M) arow1 = M - 1;
  const __bf16* Xr0 = Xb + (size_t)arow0 * K + half * 16;
  const __bf16* Xr1 = Xb + (size_t)arow1 * K + half * 16;
  const __bf16* Wc0 = Wt + (size_t)(nt0 + mlo) * K + half * 16;
  const __bf16* Wc1 = Wt + (size_t)(nt0 + 16 + mlo) * K + half * 16;

  v8f acc00 = {}, acc01 = {}, acc10 = {}, acc11 = {};

  for (int kb = 0; kb < K; kb += 32) {
    if (kb + 32 < K) {
      __builtin_prefetch(Xr0 + kb + 32, 0, 1);   // global_prefetch_b8
      __builtin_prefetch(Wc0 + kb + 32, 0, 1);
      __builtin_prefetch(Wc1 + kb + 32, 0, 1);
    }
    v8bf a0lo = *(const v8bf*)(Xr0 + kb);
    v8bf a0hi = *(const v8bf*)(Xr0 + kb + 8);
    v8bf a1lo = *(const v8bf*)(Xr1 + kb);
    v8bf a1hi = *(const v8bf*)(Xr1 + kb + 8);
    v16bf a0 = __builtin_shufflevector(a0lo, a0hi, 0,1,2,3,4,5,6,7,8,9,10,11,12,13,14,15);
    v16bf a1 = __builtin_shufflevector(a1lo, a1hi, 0,1,2,3,4,5,6,7,8,9,10,11,12,13,14,15);
    v16bf b0 = *(const v16bf*)(Wc0 + kb);
    v16bf b1 = *(const v16bf*)(Wc1 + kb);

    acc00 = __builtin_amdgcn_wmma_f32_16x16x32_bf16(false, a0, false, b0, (short)0, acc00, false, false);
    acc01 = __builtin_amdgcn_wmma_f32_16x16x32_bf16(false, a0, false, b1, (short)0, acc01, false, false);
    acc10 = __builtin_amdgcn_wmma_f32_16x16x32_bf16(false, a1, false, b0, (short)0, acc10, false, false);
    acc11 = __builtin_amdgcn_wmma_f32_16x16x32_bf16(false, a1, false, b1, (short)0, acc11, false, false);
  }

  // C/D layout: VGPR r -> M = r + 8*half, N = lane&15
  const float bias0 = bias ? bias[nt0 + mlo]      : 0.f;
  const float bias1 = bias ? bias[nt0 + 16 + mlo] : 0.f;
#pragma unroll
  for (int r = 0; r < 8; ++r) {
    int m0 = mt0 + r + half * 8;
    int m1 = mt0 + 16 + r + half * 8;
    if (m0 < M) {
      Y[(size_t)m0 * N + nt0 + mlo]      = acc00[r] + bias0;
      Y[(size_t)m0 * N + nt0 + 16 + mlo] = acc01[r] + bias1;
    }
    if (m1 < M) {
      Y[(size_t)m1 * N + nt0 + mlo]      = acc10[r] + bias0;
      Y[(size_t)m1 * N + nt0 + 16 + mlo] = acc11[r] + bias1;
    }
  }
}

// ---------------------------------------------------------------------------
// Per-row 1/||x|| for de (1024 rows) or var = nv[:,1:,:] (B*N rows).
// ---------------------------------------------------------------------------
__global__ __launch_bounds__(128) void row_inv_norm(
    const float* __restrict__ X, float* __restrict__ out, int skipCls)
{
  int r = blockIdx.x;
  const float* row;
  if (skipCls) {
    int b = r >> 6, n = r & 63;
    row = X + ((size_t)b * Ss + 1 + n) * Dd;
  } else {
    row = X + (size_t)r * Dd;
  }
  float s = 0.f;
  for (int d = threadIdx.x; d < Dd; d += 128) { float v = row[d]; s += v * v; }
  __shared__ float red[128];
  red[threadIdx.x] = s; __syncthreads();
  for (int off = 64; off > 0; off >>= 1) {
    if (threadIdx.x < off) red[threadIdx.x] += red[threadIdx.x + off];
    __syncthreads();
  }
  if (threadIdx.x == 0) out[r] = rsqrtf(red[0]);
}

// ---------------------------------------------------------------------------
// adjacency = softmax(gA * sample_sim) -> new_adj[:,1:,1:]. Block=(b,i), thr=j.
// ---------------------------------------------------------------------------
__global__ __launch_bounds__(64) void adjacency_kernel(
    const float* __restrict__ de, const float* __restrict__ nv,
    const float* __restrict__ inv_de, const float* __restrict__ inv_var,
    const float* __restrict__ topo, float* __restrict__ new_adj)
{
  int i = blockIdx.x, b = blockIdx.y, j = threadIdx.x;
  const float* dei = de + ((size_t)b * Nn + i) * Dd;
  const float* dej = de + ((size_t)b * Nn + j) * Dd;
  const float* vi  = nv + ((size_t)b * Ss + 1 + i) * Dd;
  const float* vj  = nv + ((size_t)b * Ss + 1 + j) * Dd;
  float g = 0.f, s = 0.f;
  for (int d = 0; d < Dd; d += 4) {
    float4 a = *(const float4*)(dei + d), c = *(const float4*)(dej + d);
    g += a.x * c.x + a.y * c.y + a.z * c.z + a.w * c.w;
    float4 e = *(const float4*)(vi + d), f = *(const float4*)(vj + d);
    s += e.x * f.x + e.y * f.y + e.z * f.z + e.w * f.w;
  }
  g *= inv_de[b * Nn + i] * inv_de[b * Nn + j];
  s *= inv_var[b * Nn + i] * inv_var[b * Nn + j];
  float gt = 1.f / (1.f + __expf(-(g + topo[0])));
  float gA = (gt > 0.5f) ? 1.f : 0.f;   // hard - sg(gt) + gt == hard in fwd
  if (i == j) gA = 0.f;
  float logit = gA * s;

  __shared__ float red[64];
  red[j] = logit; __syncthreads();
  float mx = -3.0e38f;
  for (int t = 0; t < 64; ++t) mx = fmaxf(mx, red[t]);
  __syncthreads();
  float e_ = __expf(logit - mx);
  red[j] = e_; __syncthreads();
  float sum = 0.f;
  for (int t = 0; t < 64; ++t) sum += red[t];
  new_adj[((size_t)b * Ss + (i + 1)) * Ss + (j + 1)] = e_ / sum;
}

__global__ __launch_bounds__(65) void adj_borders(float* __restrict__ new_adj)
{
  int b = blockIdx.x, t = threadIdx.x;
  new_adj[((size_t)b * Ss + 0) * Ss + t] = (t == 0) ? 0.f : 1.f; // row 0
  new_adj[((size_t)b * Ss + t) * Ss + 0] = 0.f;                  // col 0
}

// ---------------------------------------------------------------------------
// Fold w_edge2 (and b_edge2) with we_a = w_attn[2HD:] into W2e[D][H] (+ ch[H]).
// ---------------------------------------------------------------------------
__global__ __launch_bounds__(64) void build_w2e(
    const float* __restrict__ w_edge2, const float* __restrict__ b_edge2,
    const float* __restrict__ w_attn, float* __restrict__ W2e)
{
  int d = blockIdx.x, t = threadIdx.x;
  const float* rowp = (d < Dd) ? (w_edge2 + (size_t)d * Dd) : b_edge2;
  const float* we = w_attn + 2 * HDd;
  int h = t >> 3, seg = t & 7;
  float p = 0.f;
#pragma unroll
  for (int u = 0; u < 8; ++u) p += rowp[h * HDd + seg * 8 + u] * we[seg * 8 + u];
  __shared__ float red[64];
  red[t] = p; __syncthreads();
  if (t < 8) {
    float s = 0.f;
    for (int q = 0; q < 8; ++q) s += red[t * 8 + q];
    W2e[(size_t)d * Hh + t] = s;  // d==512 slot holds ch[h]
  }
}

// ---------------------------------------------------------------------------
// score_e[b,h,i,j] = (relu(LN(P[i']+Q[j'])) . W2e[:,h] + ch[h]) * new_adj[b,i,j]
// Grid (j-1, i, b), block 64 threads x 8 d's each.  Q already holds +b_edge1.
// ---------------------------------------------------------------------------
__global__ __launch_bounds__(64) void edge_scores(
    const float* __restrict__ P, const float* __restrict__ Q,
    const float* __restrict__ gamma, const float* __restrict__ beta,
    const float* __restrict__ W2e, const float* __restrict__ new_adj,
    float* __restrict__ score_e)
{
  int j = blockIdx.x + 1, i = blockIdx.y, b = blockIdx.z, t = threadIdx.x;
  int pi_row = (i == 0) ? (j - 1) : (i - 1);   // cls row uses concat(de_j, de_j)
  const float* Pi = P + ((size_t)b * Nn + pi_row) * Dd;
  const float* Qj = Q + ((size_t)b * Nn + (j - 1)) * Dd;

  float pre[8]; float lsum = 0.f;
  int d0 = t * 8;
#pragma unroll
  for (int u = 0; u < 8; ++u) { float v = Pi[d0 + u] + Qj[d0 + u]; pre[u] = v; lsum += v; }

  __shared__ float red[64];
  red[t] = lsum; __syncthreads();
  for (int off = 32; off > 0; off >>= 1) { if (t < off) red[t] += red[t + off]; __syncthreads(); }
  float mu = red[0] * (1.f / Dd);
  __syncthreads();
  float vs = 0.f;
#pragma unroll
  for (int u = 0; u < 8; ++u) { float dv = pre[u] - mu; vs += dv * dv; }
  red[t] = vs; __syncthreads();
  for (int off = 32; off > 0; off >>= 1) { if (t < off) red[t] += red[t + off]; __syncthreads(); }
  float rstd = rsqrtf(red[0] * (1.f / Dd) + 1e-5f);
  __syncthreads();

  float part[8] = {};
#pragma unroll
  for (int u = 0; u < 8; ++u) {
    int d = d0 + u;
    float x = (pre[u] - mu) * rstd * gamma[d] + beta[d];
    x = fmaxf(x, 0.f);
#pragma unroll
    for (int h = 0; h < 8; ++h) part[h] += x * W2e[(size_t)d * Hh + h];
  }
  __shared__ float red8[8][64];
#pragma unroll
  for (int h = 0; h < 8; ++h) red8[h][t] = part[h];
  __syncthreads();
  if (t < 8) {
    float s = W2e[(size_t)Dd * Hh + t];  // ch[h] (b_edge2 folded)
    for (int q = 0; q < 64; ++q) s += red8[t][q];
    float na = new_adj[((size_t)b * Ss + i) * Ss + j];
    score_e[(((size_t)b * Hh + t) * Ss + i) * Ss + j] = s * na;
  }
}

// ---------------------------------------------------------------------------
// score_k[b,h,j] = K[b,j,h*64:] . w_attn[HD:2HD].  One block per (b,j), wave/head.
// ---------------------------------------------------------------------------
__global__ __launch_bounds__(256) void score_k_kernel(
    const float* __restrict__ Km, const float* __restrict__ w_attn,
    float* __restrict__ sk)
{
  int row = blockIdx.x;  // b*S + j
  int h = threadIdx.x >> 5, lane = threadIdx.x & 31;
  const float* kr = Km + (size_t)row * Dd + h * HDd;
  const float* wk = w_attn + HDd;
  float s = kr[lane] * wk[lane] + kr[lane + 32] * wk[lane + 32];
  for (int off = 16; off > 0; off >>= 1) s += __shfl_xor(s, off, 32);
  if (lane == 0) {
    int b = row / Ss, j = row % Ss;
    sk[((size_t)b * Hh + h) * Ss + j] = s;
  }
}

// ---------------------------------------------------------------------------
// Fused attention softmax + context. Grid (i, b); 8 waves, wave h = head h.
// ---------------------------------------------------------------------------
__global__ __launch_bounds__(256) void attn_ctx(
    const float* __restrict__ Qm, const float* __restrict__ Vm,
    const float* __restrict__ w_attn, const float* __restrict__ b_attn,
    const float* __restrict__ sk, const float* __restrict__ score_e,
    const float* __restrict__ new_adj, float* __restrict__ attn_out,
    float* __restrict__ ctx)
{
  int i = blockIdx.x, b = blockIdx.y;
  int h = threadIdx.x >> 5, lane = threadIdx.x & 31;

  const float* qr = Qm + ((size_t)b * Ss + i) * Dd + h * HDd;
  float sq = qr[lane] * w_attn[lane] + qr[lane + 32] * w_attn[lane + 32];
  for (int off = 16; off > 0; off >>= 1) sq += __shfl_xor(sq, off, 32);

  float ba = b_attn[0];
  const float* skh = sk + ((size_t)b * Hh + h) * Ss;
  const float* seh = score_e + (((size_t)b * Hh + h) * Ss + i) * Ss;
  const float* nar = new_adj + ((size_t)b * Ss + i) * Ss;

  const float NEG = -3.0e38f;
  float lg[3];
#pragma unroll
  for (int r = 0; r < 3; ++r) {
    int j = lane + 32 * r;
    if (j < Ss) {
      float na = nar[j];
      float se = (j == 0) ? 0.f : seh[j];
      float mask = (na == 0.f) ? -1.0e9f : 0.f;
      lg[r] = sq + skh[j] + se + ba + mask;
    } else lg[r] = NEG;
  }
  float mx = fmaxf(fmaxf(lg[0], lg[1]), lg[2]);
  for (int off = 16; off > 0; off >>= 1) mx = fmaxf(mx, __shfl_xor(mx, off, 32));
  float e0 = __expf(lg[0] - mx), e1 = __expf(lg[1] - mx), e2 = __expf(lg[2] - mx);
  float sum = e0 + e1 + e2;
  for (int off = 16; off > 0; off >>= 1) sum += __shfl_xor(sum, off, 32);
  float inv = 1.f / sum;
  float p0 = e0 * inv, p1 = e1 * inv, p2 = e2 * inv;

  float* ao = attn_out + (((size_t)b * Hh + h) * Ss + i) * Ss;
  ao[lane] = p0; ao[lane + 32] = p1;
  if (lane == 0) ao[64] = p2;

  __shared__ float shp[Hh][72];
  shp[h][lane] = p0; shp[h][lane + 32] = p1;
  if (lane == 0) shp[h][64] = p2;
  __syncthreads();

  float acc0 = 0.f, acc1 = 0.f;
  const float* vb = Vm + (size_t)b * Ss * Dd + h * HDd;
  for (int j = 0; j < Ss; ++j) {
    float pj = shp[h][j];
    const float* vr = vb + (size_t)j * Dd;
    acc0 += pj * vr[lane];
    acc1 += pj * vr[lane + 32];
  }
  float* cr = ctx + ((size_t)b * Ss + i) * Dd + h * HDd;
  cr[lane] = acc0; cr[lane + 32] = acc1;
}

// ---------------------------------------------------------------------------
extern "C" void kernel_launch(void* const* d_in, const int* in_sizes, int n_in,
                              void* d_out, int out_size, void* d_ws, size_t ws_size,
                              hipStream_t stream)
{
  const float* desc   = (const float*)d_in[0];
  const float* nv     = (const float*)d_in[1];
  const float* w_gt   = (const float*)d_in[2];
  const float* b_gt   = (const float*)d_in[3];
  const float* topo   = (const float*)d_in[4];
  const float* w_e1   = (const float*)d_in[5];
  const float* b_e1   = (const float*)d_in[6];
  const float* lng    = (const float*)d_in[7];
  const float* lnb    = (const float*)d_in[8];
  const float* w_e2   = (const float*)d_in[9];
  const float* b_e2   = (const float*)d_in[10];
  const float* w_q    = (const float*)d_in[11];
  const float* b_q    = (const float*)d_in[12];
  const float* w_k    = (const float*)d_in[13];
  const float* b_k    = (const float*)d_in[14];
  const float* w_v    = (const float*)d_in[15];
  const float* b_v    = (const float*)d_in[16];
  const float* w_attn = (const float*)d_in[17];
  const float* b_attn = (const float*)d_in[18];
  const float* w_out  = (const float*)d_in[19];
  const float* b_out  = (const float*)d_in[20];

  float* out      = (float*)d_out;                       // (B,S,D)
  float* attn_out = out + (size_t)Bz * Ss * Dd;          // (B,H,S,S)

  const size_t SZ_N = (size_t)Bz * Nn * Dd;   // 524288
  const size_t SZ_S = (size_t)Bz * Ss * Dd;   // 532480
  const size_t SZ_W = (size_t)Dd * Dd;        // 262144

  // ---- fp32 workspace region ----
  float* ws = (float*)d_ws;
  float* de      = ws;
  float* Pm      = de + SZ_N;
  float* Qe      = Pm + SZ_N;
  float* qm      = Qe + SZ_N;
  float* km      = qm + SZ_S;
  float* vm      = km + SZ_S;
  float* ctx     = vm + SZ_S;
  float* inv_de  = ctx + SZ_S;                 // 1024
  float* inv_var = inv_de + Bz * Nn;           // 1024
  float* new_adj = inv_var + Bz * Nn;          // 16*65*65
  float* score_e = new_adj + (size_t)Bz * Ss * Ss;      // 16*8*65*65
  float* w2e     = score_e + (size_t)Bz * Hh * Ss * Ss; // 513*8
  float* sk      = w2e + (Dd + 1) * Hh;        // 16*8*65
  float* fend    = sk + (size_t)Bz * Hh * Ss;

  // ---- bf16 workspace region (32B-aligned: all prior sizes are /8) ----
  __bf16* bf      = (__bf16*)fend;
  __bf16* desc_bf = bf;                 // 1024*512
  __bf16* nv_bf   = desc_bf + SZ_N;     // 1040*512
  __bf16* de_bf   = nv_bf + SZ_S;       // 1024*512
  __bf16* ctx_bf  = de_bf + SZ_N;       // 1040*512
  __bf16* wt_gt   = ctx_bf + SZ_S;      // 7 x 512*512 transposed weights
  __bf16* wt_q    = wt_gt + SZ_W;
  __bf16* wt_k    = wt_q + SZ_W;
  __bf16* wt_v    = wt_k + SZ_W;
  __bf16* wt_e1a  = wt_v + SZ_W;
  __bf16* wt_e1b  = wt_e1a + SZ_W;
  __bf16* wt_out  = wt_e1b + SZ_W;

  const int MN = Bz * Nn;   // 1024
  const int MS = Bz * Ss;   // 1040

  dim3 gN(Dd / 32, (MN + 31) / 32);
  dim3 gS(Dd / 32, (MS + 31) / 32);
  dim3 gT(Dd / 32, Dd / 32);
  dim3 bT(32, 8);

  // 0) one-time precision/layout conversion
  to_bf16<<<(SZ_N / 4 + 255) / 256, 256, 0, stream>>>(desc, desc_bf, (int)(SZ_N / 4));
  to_bf16<<<(SZ_S / 4 + 255) / 256, 256, 0, stream>>>(nv, nv_bf, (int)(SZ_S / 4));
  transpose_to_bf16<<<gT, bT, 0, stream>>>(w_gt, wt_gt);
  transpose_to_bf16<<<gT, bT, 0, stream>>>(w_q,  wt_q);
  transpose_to_bf16<<<gT, bT, 0, stream>>>(w_k,  wt_k);
  transpose_to_bf16<<<gT, bT, 0, stream>>>(w_v,  wt_v);
  transpose_to_bf16<<<gT, bT, 0, stream>>>(w_e1,            wt_e1a);
  transpose_to_bf16<<<gT, bT, 0, stream>>>(w_e1 + SZ_W,     wt_e1b);
  transpose_to_bf16<<<gT, bT, 0, stream>>>(w_out, wt_out);

  // 1) de = desc @ w_gt + b_gt
  wmma_gemm_bias<<<gN, 32, 0, stream>>>(desc_bf, wt_gt, b_gt, de, MN, Dd, Dd);
  to_bf16<<<(SZ_N / 4 + 255) / 256, 256, 0, stream>>>(de, de_bf, (int)(SZ_N / 4));
  // 2) q/k/v = nv @ w_{q,k,v} + b
  wmma_gemm_bias<<<gS, 32, 0, stream>>>(nv_bf, wt_q, b_q, qm, MS, Dd, Dd);
  wmma_gemm_bias<<<gS, 32, 0, stream>>>(nv_bf, wt_k, b_k, km, MS, Dd, Dd);
  wmma_gemm_bias<<<gS, 32, 0, stream>>>(nv_bf, wt_v, b_v, vm, MS, Dd, Dd);
  // 3) P = de @ w_edge1[:D];  Q = de @ w_edge1[D:] + b_edge1
  wmma_gemm_bias<<<gN, 32, 0, stream>>>(de_bf, wt_e1a, nullptr, Pm, MN, Dd, Dd);
  wmma_gemm_bias<<<gN, 32, 0, stream>>>(de_bf, wt_e1b, b_e1, Qe, MN, Dd, Dd);
  // 4) row inverse norms
  row_inv_norm<<<MN, 128, 0, stream>>>(de, inv_de, 0);
  row_inv_norm<<<MN, 128, 0, stream>>>(nv, inv_var, 1);
  // 5) adjacency softmax -> new_adj interior; borders
  adjacency_kernel<<<dim3(Nn, Bz), 64, 0, stream>>>(de, nv, inv_de, inv_var, topo, new_adj);
  adj_borders<<<Bz, Ss, 0, stream>>>(new_adj);
  // 6) fold edge layer 2 with we_a
  build_w2e<<<Dd + 1, 64, 0, stream>>>(w_e2, b_e2, w_attn, w2e);
  // 7) fused LN+ReLU+edge score
  edge_scores<<<dim3(Nn, Ss, Bz), 64, 0, stream>>>(Pm, Qe, lng, lnb, w2e, new_adj, score_e);
  // 8) score_k
  score_k_kernel<<<MS, 256, 0, stream>>>(km, w_attn, sk);
  // 9) attention softmax + context (writes attn_out)
  attn_ctx<<<dim3(Ss, Bz), 256, 0, stream>>>(qm, vm, w_attn, b_attn, sk, score_e, new_adj, attn_out, ctx);
  // 10) output = ctx @ w_out + b_out
  to_bf16<<<(SZ_S / 4 + 255) / 256, 256, 0, stream>>>(ctx, ctx_bf, (int)(SZ_S / 4));
  wmma_gemm_bias<<<gS, 32, 0, stream>>>(ctx_bf, wt_out, b_out, out, MS, Dd, Dd);
}
